// SparseMoE_43224550867277
// MI455X (gfx1250) — compile-verified, matrix operand order checked by reading, not run on hardware
//
#include <hip/hip_runtime.h>
#include <math.h>

#define D_MODEL 1024
#define D_FF    4096
#define N_EXP   8
#define M_TILE  32
#define FF_CHUNK 128
#define RS (D_MODEL + 8)   // padded bf16 row stride for x tile (elements)
#define HS (FF_CHUNK + 8)  // padded bf16 row stride for H tile (elements)
#define KS1 68             // [n][k] stride for staged W1 block (64 k + pad), bytes*2 % 8 == 0
#define KS2 36             // [n][k] stride for staged W2 block (32 k + pad), bytes*2 % 8 == 0

typedef __attribute__((ext_vector_type(16))) __bf16 v16bf;
typedef __attribute__((ext_vector_type(8)))  float  v8f;

union ABFrag { unsigned int u[8]; v16bf v; };

// single-instruction packed fp32->bf16 (round-to-nearest-even)
__device__ __forceinline__ unsigned int pk2bf(float lo, float hi) {
    unsigned int r;
    asm("v_cvt_pk_bf16_f32 %0, %1, %2" : "=v"(r) : "v"(lo), "v"(hi));
    return r;
}
__device__ __forceinline__ unsigned short f2bf(float f) {
    return (unsigned short)(pk2bf(f, f) & 0xFFFFu);
}

// ---------------- kernel 0: zero output + expert counters ----------------
__global__ void moe_zero(float* __restrict__ out, size_t n, int* __restrict__ cnt) {
    size_t i = (size_t)blockIdx.x * blockDim.x + threadIdx.x;
    size_t stride = (size_t)gridDim.x * blockDim.x;
    for (; i < n; i += stride) out[i] = 0.0f;
    if (blockIdx.x == 0 && threadIdx.x < N_EXP) cnt[threadIdx.x] = 0;
}

// ---------------- kernel 1: router (gate GEMV + top-2 softmax + scatter) ----------------
__global__ void moe_router(const float* __restrict__ x, const float* __restrict__ gw,
                           int* __restrict__ cnt, int* __restrict__ btok,
                           float* __restrict__ bwt, int ntok) {
    __shared__ float sg[D_MODEL * N_EXP];           // 32 KB
    for (int i = threadIdx.x; i < D_MODEL * N_EXP; i += blockDim.x) sg[i] = gw[i];
    __syncthreads();

    int t = blockIdx.x * blockDim.x + threadIdx.x;
    if (t >= ntok) return;

    float acc[N_EXP];
    #pragma unroll
    for (int e = 0; e < N_EXP; ++e) acc[e] = 0.0f;

    const float* xr = x + (size_t)t * D_MODEL;
    for (int d = 0; d < D_MODEL; ++d) {
        float xv = xr[d];
        #pragma unroll
        for (int e = 0; e < N_EXP; ++e) acc[e] += xv * sg[d * N_EXP + e];
    }

    // top-2, lowest index wins ties (matches jax.lax.top_k)
    int i0 = 0; float v0 = acc[0];
    #pragma unroll
    for (int e = 1; e < N_EXP; ++e) if (acc[e] > v0) { v0 = acc[e]; i0 = e; }
    int i1 = -1; float v1 = -3.4e38f;
    #pragma unroll
    for (int e = 0; e < N_EXP; ++e) {
        if (e == i0) continue;
        if (acc[e] > v1) { v1 = acc[e]; i1 = e; }
    }
    float ex = __expf(v1 - v0);
    float denom = 1.0f + ex;
    float w0 = 1.0f / denom;
    float w1 = ex / denom;

    int s0 = atomicAdd(&cnt[i0], 1);
    btok[i0 * ntok + s0] = t;  bwt[i0 * ntok + s0] = w0;
    int s1 = atomicAdd(&cnt[i1], 1);
    btok[i1 * ntok + s1] = t;  bwt[i1 * ntok + s1] = w1;
}

// ---------------- kernel 2: fused expert FFN (bf16 WMMA, LDS-staged weights) ----------------
__global__ __launch_bounds__(512, 1) void moe_ffn(
    const float* __restrict__ x,  const float* __restrict__ W1, const float* __restrict__ b1,
    const float* __restrict__ W2, const float* __restrict__ b2,
    const int* __restrict__ cnt,  const int* __restrict__ btok, const float* __restrict__ bwt,
    float* __restrict__ out, int ntok) {

    const int e = blockIdx.y;
    const int count = cnt[e];
    const int row0 = blockIdx.x * M_TILE;
    if (row0 >= count) return;

    __shared__ unsigned short sA[M_TILE * RS];     // x tile bf16        (~66 KB)
    __shared__ unsigned short sH[M_TILE * HS];     // H chunk bf16       (~8.7 KB)
    __shared__ unsigned short sB[KS2 * D_MODEL];   // staged weights, [n][k] transposed (~72 KB)
    __shared__ int   sTok[M_TILE];
    __shared__ float sW[M_TILE];

    const int t = threadIdx.x;

    if (t < M_TILE) {
        int idx = row0 + t;
        if (idx < count) { sTok[t] = btok[e * ntok + idx]; sW[t] = bwt[e * ntok + idx]; }
        else             { sTok[t] = -1;                   sW[t] = 0.0f; }
    }
    __syncthreads();

    // gather + convert x tile to bf16 (each thread: 64 floats = 16 float4 of one row)
    {
        int r  = t >> 4;                     // 0..31
        int c0 = (t & 15) * (D_MODEL / 16);  // 64-float chunk
        int tok = sTok[r];
        unsigned int* dst = (unsigned int*)&sA[r * RS + c0];
        if (tok >= 0) {
            const float4* src = (const float4*)(x + (size_t)tok * D_MODEL + c0);
            #pragma unroll
            for (int i = 0; i < 16; ++i) {
                float4 v = src[i];
                dst[2 * i]     = pk2bf(v.x, v.y);
                dst[2 * i + 1] = pk2bf(v.z, v.w);
            }
        } else {
            #pragma unroll
            for (int i = 0; i < 32; ++i) dst[i] = 0u;
        }
    }

    const int wid  = t >> 5;
    const int lane = t & 31;
    const int mh   = wid >> 3;       // 0..1 : M half (16 rows)
    const int nq   = wid & 7;        // 0..7 : N quadrant
    const int hi   = lane >> 4;
    const int lm   = lane & 15;

    // K pair-offsets for bf16 A/B fragments (ISA 7.12.2 16-bit layout)
    int koff[8];
    #pragma unroll
    for (int j = 0; j < 8; ++j)
        koff[j] = ((j < 4) ? (2 * j) : (16 + 2 * (j - 4))) + (hi ? 8 : 0);

    v8f acc2[8];
    #pragma unroll
    for (int nt = 0; nt < 8; ++nt)
        #pragma unroll
        for (int v = 0; v < 8; ++v) acc2[nt][v] = 0.0f;

    const float* W1e = W1 + (size_t)e * D_MODEL * D_FF;
    const float* W2e = W2 + (size_t)e * D_FF * D_MODEL;

    for (int ff0 = 0; ff0 < D_FF; ff0 += FF_CHUNK) {
        // ======== GEMM1: H tile = x @ W1[:, ff0:ff0+128], K = 1024 ========
        v8f c1;
        #pragma unroll
        for (int v = 0; v < 8; ++v) c1[v] = 0.0f;

        for (int k0 = 0; k0 < D_MODEL; k0 += 64) {
            __syncthreads();   // protect sB from previous readers
            // stage W1 block [k0:k0+64, ff0:ff0+128] -> sB[n][k] (bf16, K-pairs packed)
            #pragma unroll
            for (int i = 0; i < 2; ++i) {
                int a  = t + i * 512;        // 0..1023
                int kp = a >> 5;             // 0..31  (pair of k)
                int n4 = a & 31;             // 0..31  (group of 4 n)
                const float4* r0 = (const float4*)(W1e + (size_t)(k0 + 2 * kp)     * D_FF + ff0 + n4 * 4);
                const float4* r1 = (const float4*)(W1e + (size_t)(k0 + 2 * kp + 1) * D_FF + ff0 + n4 * 4);
                float4 v0 = *r0, v1 = *r1;
                *(unsigned int*)&sB[(n4 * 4 + 0) * KS1 + 2 * kp] = pk2bf(v0.x, v1.x);
                *(unsigned int*)&sB[(n4 * 4 + 1) * KS1 + 2 * kp] = pk2bf(v0.y, v1.y);
                *(unsigned int*)&sB[(n4 * 4 + 2) * KS1 + 2 * kp] = pk2bf(v0.z, v1.z);
                *(unsigned int*)&sB[(n4 * 4 + 3) * KS1 + 2 * kp] = pk2bf(v0.w, v1.w);
            }
            __syncthreads();

            const int nl = nq * 16 + lm;     // this lane's local B column
            #pragma unroll
            for (int ks = 0; ks < 64; ks += 32) {
                ABFrag fa, fb;
                const unsigned short* arow = &sA[(mh * 16 + lm) * RS + k0 + ks];
                #pragma unroll
                for (int j = 0; j < 8; ++j)
                    fa.u[j] = *(const unsigned int*)(arow + koff[j]);
                #pragma unroll
                for (int j = 0; j < 8; ++j)
                    fb.u[j] = *(const unsigned int*)&sB[nl * KS1 + ks + koff[j]];
                c1 = __builtin_amdgcn_wmma_f32_16x16x32_bf16(false, fa.v, false, fb.v,
                                                             (short)0, c1, false, false);
            }
        }

        // bias + exact GELU + bf16 store to H chunk
        {
            float bias1 = b1[e * D_FF + ff0 + nq * 16 + lm];
            #pragma unroll
            for (int v = 0; v < 8; ++v) {
                float h = c1[v] + bias1;
                h = 0.5f * h * (1.0f + erff(h * 0.70710678118654752f));
                int row = mh * 16 + v + hi * 8;
                sH[row * HS + nq * 16 + lm] = f2bf(h);
            }
        }

        // ======== GEMM2: acc2 += H_chunk @ W2[ff0:ff0+128, :] ========
        #pragma unroll
        for (int kt = 0; kt < FF_CHUNK / 32; ++kt) {
            __syncthreads();   // sH ready / protect sB
            // stage W2 block [ff0+kt*32 : +32, 0:1024] -> sB[n][k]
            const int ffk = ff0 + kt * 32;
            #pragma unroll
            for (int i = 0; i < 8; ++i) {
                int a  = t + i * 512;        // 0..4095
                int kp = a >> 8;             // 0..15 (pair of k)
                int n4 = a & 255;            // 0..255 (group of 4 n)
                const float4* r0 = (const float4*)(W2e + (size_t)(ffk + 2 * kp)     * D_MODEL + n4 * 4);
                const float4* r1 = (const float4*)(W2e + (size_t)(ffk + 2 * kp + 1) * D_MODEL + n4 * 4);
                float4 v0 = *r0, v1 = *r1;
                *(unsigned int*)&sB[(n4 * 4 + 0) * KS2 + 2 * kp] = pk2bf(v0.x, v1.x);
                *(unsigned int*)&sB[(n4 * 4 + 1) * KS2 + 2 * kp] = pk2bf(v0.y, v1.y);
                *(unsigned int*)&sB[(n4 * 4 + 2) * KS2 + 2 * kp] = pk2bf(v0.z, v1.z);
                *(unsigned int*)&sB[(n4 * 4 + 3) * KS2 + 2 * kp] = pk2bf(v0.w, v1.w);
            }
            __syncthreads();

            ABFrag fa;
            const unsigned short* hrow = &sH[(mh * 16 + lm) * HS + kt * 32];
            #pragma unroll
            for (int j = 0; j < 8; ++j)
                fa.u[j] = *(const unsigned int*)(hrow + koff[j]);
            #pragma unroll
            for (int nt = 0; nt < 8; ++nt) {
                ABFrag fb;
                const int col = nq * 128 + nt * 16 + lm;
                #pragma unroll
                for (int j = 0; j < 8; ++j)
                    fb.u[j] = *(const unsigned int*)&sB[col * KS2 + koff[j]];
                acc2[nt] = __builtin_amdgcn_wmma_f32_16x16x32_bf16(false, fa.v, false, fb.v,
                                                                   (short)0, acc2[nt], false, false);
            }
        }
        __syncthreads();
    }

    // ---- epilogue: + b2, scale by routing weight, atomic combine ----
    #pragma unroll
    for (int nt = 0; nt < 8; ++nt) {
        int col = nq * 128 + nt * 16 + lm;
        float bias2 = b2[e * D_MODEL + col];
        #pragma unroll
        for (int v = 0; v < 8; ++v) {
            int row = mh * 16 + v + hi * 8;
            int tok = sTok[row];
            if (tok >= 0) {
                float val = (acc2[nt][v] + bias2) * sW[row];
                atomicAdd(&out[(size_t)tok * D_MODEL + col], val);
            }
        }
    }
}

// ---------------- launch ----------------
extern "C" void kernel_launch(void* const* d_in, const int* in_sizes, int n_in,
                              void* d_out, int out_size, void* d_ws, size_t ws_size,
                              hipStream_t stream) {
    const float* x   = (const float*)d_in[0];
    const float* gw  = (const float*)d_in[1];
    const float* W1  = (const float*)d_in[2];
    const float* b1  = (const float*)d_in[3];
    const float* W2  = (const float*)d_in[4];
    const float* b2  = (const float*)d_in[5];
    float* out = (float*)d_out;

    const int ntok = in_sizes[0] / D_MODEL;   // 4096

    // workspace layout: [cnt: 8 ints][pad to 256B][btok: E*ntok ints][bwt: E*ntok floats]
    char* ws = (char*)d_ws;
    int*   cnt  = (int*)ws;
    int*   btok = (int*)(ws + 256);
    float* bwt  = (float*)(ws + 256 + sizeof(int) * (size_t)N_EXP * ntok);

    moe_zero<<<1024, 256, 0, stream>>>(out, (size_t)out_size, cnt);
    moe_router<<<(ntok + 255) / 256, 256, 0, stream>>>(x, gw, cnt, btok, bwt, ntok);

    int tiles = (ntok + M_TILE - 1) / M_TILE;
    dim3 grid(tiles, N_EXP);
    moe_ffn<<<grid, 512, 0, stream>>>(x, W1, b1, W2, b2, cnt, btok, bwt, out, ntok);
}
